// AttentionWindow_75909251990144
// MI455X (gfx1250) — matrix operand order repaired and checked by reading, not compiled, not run on hardware
//
#include <hip/hip_runtime.h>
#include <hip/hip_bf16.h>
#include <stdint.h>

// Problem constants (match reference)
#define BATCH   8
#define SEQ     2048
#define HIDDEN  1024
#define WSIZE   64
#define NEG_INF (-1e9f)

typedef __attribute__((ext_vector_type(16))) __bf16        v16bf;
typedef __attribute__((ext_vector_type(8)))  float         v8f;
typedef __attribute__((ext_vector_type(4)))  unsigned int  v4u;
typedef __attribute__((ext_vector_type(8)))  int           v8i;
typedef __attribute__((ext_vector_type(4)))  int           v4i;

union AFrag {
    v16bf    v;
    uint32_t u[8];
    uint16_t h[16];
};

__device__ __forceinline__ uint16_t f32_to_bf16(float f) {
    uint32_t u = __float_as_uint(f);
    uint32_t r = u + 0x7FFFu + ((u >> 16) & 1u);   // round-to-nearest-even
    return (uint16_t)(r >> 16);
}

// Low 32 bits of a flat shared-aperture address = LDS byte offset (ISA 10.2).
__device__ __forceinline__ uint32_t lds_offset(const void* p) {
    return (uint32_t)(uintptr_t)p;
}

// ---------------------------------------------------------------------------
// Tensor Data Mover: 2-D tile load, bf16 elements (data_size=2B).
// D# packed per CDNA5 ISA 8.3/8.4: group0 = {flags, lds_addr, global_addr,
// type=2}; group1 = {data_size, tensor_dim0/1, tile_dim0/1, dim0 stride}.
// Groups 2/3 zero (2-D tile; tile_dim2=0 -> unused).
// This toolchain exposes the 6-arg builtin (g0, g1, g2, g3, g4, cpol).
// ---------------------------------------------------------------------------
__device__ __forceinline__ void tdm_load_2d(uint32_t lds_off, const void* gptr,
                                            uint32_t td0, uint32_t td1,
                                            uint32_t tl0, uint32_t tl1,
                                            uint32_t stride0) {
    uint64_t ga = (uint64_t)(uintptr_t)gptr;
    v4u g0;
    g0[0] = 1u;                                          // count=1 (valid), user mode
    g0[1] = lds_off;                                     // LDS byte address
    g0[2] = (uint32_t)ga;                                // global_addr[31:0]
    g0[3] = (uint32_t)((ga >> 32) & 0x01FFFFFFull)       // global_addr[56:32]
          | (2u << 30);                                  // type=2 ("image")
    v8i g1;
    g1[0] = (int)(1u << 16);                             // data_size=1 -> 2 bytes
    g1[1] = (int)((td0 & 0xFFFFu) << 16);                // tensor_dim0[15:0]
    g1[2] = (int)((td0 >> 16) | ((td1 & 0xFFFFu) << 16));// dim0 hi | dim1 lo
    g1[3] = (int)((td1 >> 16) | ((tl0 & 0xFFFFu) << 16));// dim1 hi | tile_dim0
    g1[4] = (int)(tl1 & 0xFFFFu);                        // tile_dim1 (tile_dim2=0)
    g1[5] = (int)stride0;                                // tensor_dim0_stride lo32
    g1[6] = 0;
    g1[7] = 0;
    v4i z4 = {0, 0, 0, 0};
    v8i z8 = {0, 0, 0, 0, 0, 0, 0, 0};
    __builtin_amdgcn_tensor_load_to_lds(g0, g1, z4, z4, z8, 0);
}

// ---------------------------------------------------------------------------
// Stage 0a: f32 -> bf16 conversion (grid-stride)
// ---------------------------------------------------------------------------
__global__ void cvt_bf16_kernel(const float* __restrict__ src,
                                uint16_t* __restrict__ dst, int n) {
    int i = blockIdx.x * blockDim.x + threadIdx.x;
    int stride = gridDim.x * blockDim.x;
    for (; i < n; i += stride) dst[i] = f32_to_bf16(src[i]);
}

// ---------------------------------------------------------------------------
// Stage 0b: Wt[n][k] = bf16(W[k][n])  (tiled transpose, 32x32 tiles)
// ---------------------------------------------------------------------------
__global__ void __launch_bounds__(256)
transpose_w_kernel(const float* __restrict__ src, uint16_t* __restrict__ dst) {
    __shared__ uint16_t t[32][33];
    const int bx = blockIdx.x * 32;   // n
    const int by = blockIdx.y * 32;   // k
    for (int i = threadIdx.y; i < 32; i += 8)
        t[i][threadIdx.x] = f32_to_bf16(src[(size_t)(by + i) * HIDDEN + bx + threadIdx.x]);
    __syncthreads();
    for (int i = threadIdx.y; i < 32; i += 8)
        dst[(size_t)(bx + i) * HIDDEN + by + threadIdx.x] = t[threadIdx.x][i];
}

// ---------------------------------------------------------------------------
// Stage 0c: reprT[b][d][s] = bf16(repr[b][s][d])  (tiled transpose per batch)
// ---------------------------------------------------------------------------
__global__ void __launch_bounds__(256)
transpose_repr_kernel(const float* __restrict__ src, uint16_t* __restrict__ dst) {
    __shared__ uint16_t t[32][33];
    const int b  = blockIdx.z;
    const float*    s = src + (size_t)b * SEQ * HIDDEN;
    uint16_t*       d = dst + (size_t)b * SEQ * HIDDEN;
    const int dx = blockIdx.x * 32;   // hidden dim
    const int sy = blockIdx.y * 32;   // seq dim
    for (int i = threadIdx.y; i < 32; i += 8)
        t[i][threadIdx.x] = f32_to_bf16(s[(size_t)(sy + i) * HIDDEN + dx + threadIdx.x]);
    __syncthreads();
    for (int i = threadIdx.y; i < 32; i += 8)
        d[(size_t)(dx + i) * SEQ + sy + threadIdx.x] = t[threadIdx.x][i];
}

// ---------------------------------------------------------------------------
// Stage 1: P = repr @ W  (bf16 in, f32 accumulate, bf16 out)
// 128 threads (4 waves) per 64x64 tile; A/B chunks staged via TDM.
// ---------------------------------------------------------------------------
__global__ void __launch_bounds__(128)
gemm_rw_kernel(const uint16_t* __restrict__ A,    // [BATCH*SEQ][HIDDEN] bf16
               const uint16_t* __restrict__ Wt,   // [HIDDEN][HIDDEN] bf16, Wt[n][k]
               uint16_t* __restrict__ P) {        // [BATCH*SEQ][HIDDEN] bf16
    __shared__ __align__(16) uint16_t As[64][32];   // As[m][k]
    __shared__ __align__(16) uint16_t Bs[64][32];   // Bs[n][k]

    const int tid  = threadIdx.x;
    const int wave = tid >> 5;
    const int lane = tid & 31;
    const int hi   = (lane >> 4) & 1;
    const int l15  = lane & 15;
    const int M0   = blockIdx.x * 64;
    const int N0   = blockIdx.y * 64;

    const v8f vzero = {0.f,0.f,0.f,0.f,0.f,0.f,0.f,0.f};
    v8f acc[4];
    acc[0] = vzero; acc[1] = vzero; acc[2] = vzero; acc[3] = vzero;

    for (int kc = 0; kc < HIDDEN; kc += 32) {
        if (wave == 0) {   // wave-uniform; TDM ignores EXEC
            tdm_load_2d(lds_offset(&As[0][0]),
                        A + (size_t)M0 * HIDDEN + kc,
                        HIDDEN, BATCH * SEQ, 32, 64, HIDDEN);
            tdm_load_2d(lds_offset(&Bs[0][0]),
                        Wt + (size_t)N0 * HIDDEN + kc,
                        HIDDEN, HIDDEN, 32, 64, HIDDEN);
            __builtin_amdgcn_s_wait_tensorcnt(0);
        }
        __syncthreads();

        AFrag a;   // contiguous k order chosen here (same as B): VGPR v pair covers K = 2*v + hi*16
        {
            int m = wave * 16 + l15;
#pragma unroll
            for (int v = 0; v < 8; v++) {
                int k = v * 2 + hi * 16;
                a.u[v] = *(const uint32_t*)&As[m][k];
            }
        }
#pragma unroll
        for (int ns = 0; ns < 4; ns++) {
            AFrag bf;
            int n  = ns * 16 + l15;
            int k0 = hi * 16;          // same contiguous k order as A: K = elem + hi*16
            *(uint4*)&bf.h[0] = *(const uint4*)&Bs[n][k0];
            *(uint4*)&bf.h[8] = *(const uint4*)&Bs[n][k0 + 8];
            acc[ns] = __builtin_amdgcn_wmma_f32_16x16x32_bf16(
                false, a.v, false, bf.v, (short)0, acc[ns], false, false);
        }
        __syncthreads();
    }

#pragma unroll
    for (int ns = 0; ns < 4; ns++) {
#pragma unroll
        for (int r = 0; r < 8; r++) {
            int m = M0 + wave * 16 + r + hi * 8;
            int n = N0 + ns * 16 + l15;
            P[(size_t)m * HIDDEN + n] = f32_to_bf16(acc[ns][r]);
        }
    }
}

// ---------------------------------------------------------------------------
// Stage 2: banded attention. One block (4 waves) per 16-query tile.
// Window keys: j in [q0-64, q0+79] -> 144 keys, padded to 160.
// ---------------------------------------------------------------------------
__device__ __forceinline__ AFrag load_key_bfrag(const uint16_t* __restrict__ reprB,
                                                size_t rowBase, int j, int kcol,
                                                int hi) {
    AFrag bf;
    if (j >= 0 && j < SEQ) {
        const uint16_t* g = reprB + (rowBase + (size_t)j) * HIDDEN + kcol + hi * 16;
        *(uint4*)&bf.h[0] = *(const uint4*)g;
        *(uint4*)&bf.h[8] = *(const uint4*)(g + 8);
    } else {
#pragma unroll
        for (int v = 0; v < 8; v++) bf.u[v] = 0u;
    }
    return bf;
}

__global__ void __launch_bounds__(128)
attn_kernel(const float*    __restrict__ reprF,   // [B][S][D] f32
            const uint16_t* __restrict__ reprB,   // [B][S][D] bf16
            const uint16_t* __restrict__ reprT,   // [B][D][S] bf16 (transposed)
            const uint16_t* __restrict__ Pm,      // [B][S][D] bf16 (repr @ W)
            float*          __restrict__ out) {   // [B][S][D] f32
    // LDS arena (48128 B): [0..32768) Ptile[16][1024]  (phase A)
    //                      [0..20480) Vt[64][160]      (phase C, aliases Ptile)
    //                      [32768..43008) S[16][160] f32
    //                      [43008..48128) Pr[16][160] bf16
    __shared__ __align__(16) uint8_t smem[48128];
    uint16_t (*Ptile)[1024] = (uint16_t(*)[1024])smem;
    uint16_t (*Vt)[160]     = (uint16_t(*)[160])smem;
    float    (*S)[160]      = (float(*)[160])(smem + 32768);
    uint16_t (*Pr)[160]     = (uint16_t(*)[160])(smem + 43008);

    const int tid  = threadIdx.x;
    const int wave = tid >> 5;
    const int lane = tid & 31;
    const int hi   = (lane >> 4) & 1;
    const int l15  = lane & 15;
    const int b    = blockIdx.x >> 7;     // / (SEQ/16)
    const int tile = blockIdx.x & 127;
    const int q0   = tile * 16;
    const int kb   = q0 - WSIZE;          // window key base (may be < 0)
    const size_t rowBase = (size_t)b * SEQ;

    const v8f vzero = {0.f,0.f,0.f,0.f,0.f,0.f,0.f,0.f};

    // ---- TDM: stage the 16x1024 P tile into LDS once ----
    if (wave == 0) {
        tdm_load_2d(lds_offset(&Ptile[0][0]),
                    Pm + (rowBase + q0) * HIDDEN,
                    HIDDEN, BATCH * SEQ, 1024, 16, HIDDEN);
        __builtin_amdgcn_s_wait_tensorcnt(0);
    }
    __syncthreads();

    // ---- Phase A: scores = P · repr^T over the 144-key band ----
    // key tiles 0..8; wave w owns kt = w, w+4 (+ kt=8 on wave 0)
    v8f acc0 = vzero, acc1 = vzero, acc2 = vzero;

    for (int kc = 0; kc < HIDDEN; kc += 32) {
        AFrag a;
#pragma unroll
        for (int v = 0; v < 8; v++) {
            int k = kc + v * 2 + hi * 16;
            a.u[v] = *(const uint32_t*)&Ptile[l15][k];
        }
        if (kc + 32 < HIDDEN) {   // prefetch next K-chunk of this wave's key row
            int jp = kb + wave * 16 + l15;
            if (jp >= 0 && jp < SEQ)
                __builtin_prefetch(reprB + (rowBase + jp) * HIDDEN + kc + 32);
        }
        {
            AFrag bf = load_key_bfrag(reprB, rowBase, kb + wave * 16 + l15, kc, hi);
            acc0 = __builtin_amdgcn_wmma_f32_16x16x32_bf16(
                false, a.v, false, bf.v, (short)0, acc0, false, false);
        }
        {
            AFrag bf = load_key_bfrag(reprB, rowBase, kb + (wave + 4) * 16 + l15, kc, hi);
            acc1 = __builtin_amdgcn_wmma_f32_16x16x32_bf16(
                false, a.v, false, bf.v, (short)0, acc1, false, false);
        }
        if (wave == 0) {   // wave-uniform branch: EXEC stays all-1s inside
            AFrag bf = load_key_bfrag(reprB, rowBase, kb + 128 + l15, kc, hi);
            acc2 = __builtin_amdgcn_wmma_f32_16x16x32_bf16(
                false, a.v, false, bf.v, (short)0, acc2, false, false);
        }
    }

    // masked score write: D layout M = r + hi*8, N = l15
    {
#pragma unroll
        for (int t = 0; t < 2; t++) {
            int kt = wave + 4 * t;
            v8f av = (t == 0) ? acc0 : acc1;
#pragma unroll
            for (int r = 0; r < 8; r++) {
                int m  = r + hi * 8;
                int n  = kt * 16 + l15;
                int iq = q0 + m;
                int j  = kb + n;
                int d  = iq - j;
                bool masked = (j < 0) | (j >= SEQ) | (d > WSIZE) | (d < -WSIZE);
                S[m][n] = masked ? NEG_INF : av[r];
            }
        }
        if (wave == 0) {
#pragma unroll
            for (int r = 0; r < 8; r++) {
                int m  = r + hi * 8;
                int n  = 128 + l15;
                int iq = q0 + m;
                int j  = kb + n;
                int d  = iq - j;
                bool masked = (j < 0) | (j >= SEQ) | (d > WSIZE) | (d < -WSIZE);
                S[m][n] = masked ? NEG_INF : acc2[r];
            }
        }
    }
    __syncthreads();

    // ---- Phase B: per-row softmax over the 144-key band ----
    if (tid < 16) {
        int row = tid;
        float mx = NEG_INF;
        for (int c = 0; c < 144; c++) mx = fmaxf(mx, S[row][c]);
        float sum = 0.f;
        for (int c = 0; c < 144; c++) {
            float e = __expf(S[row][c] - mx);
            S[row][c] = e;
            sum += e;
        }
        float inv = 1.0f / sum;   // diagonal always unmasked -> sum >= exp(0)
        for (int c = 0; c < 144; c++) Pr[row][c] = f32_to_bf16(S[row][c] * inv);
        for (int c = 144; c < 160; c++) Pr[row][c] = 0;
    }
    __syncthreads();

    // ---- probs A-fragments for K = 0..159 (5 chunks), reused for all N-tiles
    AFrag pa[5];
#pragma unroll
    for (int kc = 0; kc < 5; kc++) {
#pragma unroll
        for (int v = 0; v < 8; v++) {
            int k = kc * 32 + v * 2 + hi * 16;
            pa[kc].u[v] = *(const uint32_t*)&Pr[l15][k];
        }
    }

    // ---- Phase C: ra = probs @ V, strips of 64 hidden cols; fused epilogue
    const bool interior = (kb >= 0) && (kb + 160 <= SEQ);
    for (int it = 0; it < 16; it++) {
        const int c0 = it * 64;
        if (interior) {
            // TDM: [64 cols x 160 keys] strip straight from pre-transposed reprT
            if (wave == 0) {
                tdm_load_2d(lds_offset(&Vt[0][0]),
                            reprT + (size_t)b * HIDDEN * SEQ + (size_t)c0 * SEQ + kb,
                            SEQ, HIDDEN, 160, 64, SEQ);
                __builtin_amdgcn_s_wait_tensorcnt(0);
            }
        } else {
            // boundary tiles: manual masked transpose-staging from reprB
            for (int s = 0; s < 5; s++) {
                int idx = tid + 128 * s;        // 0..639 -> (key, colseg)
                int kk  = idx >> 2;             // 0..159
                int cs  = (idx & 3) * 16;       // 0,16,32,48
                int j   = kb + kk;
                uint16_t tmp[16];
                if (j >= 0 && j < SEQ) {
                    const uint16_t* g = reprB + (rowBase + (size_t)j) * HIDDEN + c0 + cs;
                    *(uint4*)&tmp[0] = *(const uint4*)g;
                    *(uint4*)&tmp[8] = *(const uint4*)(g + 8);
                } else {
#pragma unroll
                    for (int c = 0; c < 16; c++) tmp[c] = 0;
                }
#pragma unroll
                for (int c = 0; c < 16; c++) Vt[cs + c][kk] = tmp[c];
            }
        }
        __syncthreads();

        v8f c = vzero;
#pragma unroll
        for (int kc = 0; kc < 5; kc++) {
            AFrag bf;
            int col = wave * 16 + l15;         // N = lane&15 within subtile
            int k0  = kc * 32 + hi * 16;       // K = elem + hi*16
            *(uint4*)&bf.h[0] = *(const uint4*)&Vt[col][k0];
            *(uint4*)&bf.h[8] = *(const uint4*)&Vt[col][k0 + 8];
            c = __builtin_amdgcn_wmma_f32_16x16x32_bf16(
                false, pa[kc].v, false, bf.v, (short)0, c, false, false);
        }
        // out = repr + relu(ra)
#pragma unroll
        for (int r = 0; r < 8; r++) {
            int m = r + hi * 8;
            int n = c0 + wave * 16 + l15;
            size_t o = (rowBase + q0 + m) * HIDDEN + n;
            float ra = c[r];
            out[o] = reprF[o] + (ra > 0.f ? ra : 0.f);
        }
        __syncthreads();   // protect Vt before next strip's staging
    }
}

// ---------------------------------------------------------------------------
// Launch
// ---------------------------------------------------------------------------
extern "C" void kernel_launch(void* const* d_in, const int* in_sizes, int n_in,
                              void* d_out, int out_size, void* d_ws, size_t ws_size,
                              hipStream_t stream) {
    (void)in_sizes; (void)n_in; (void)out_size; (void)ws_size;

    const float* repr = (const float*)d_in[0];   // [8,2048,1024] f32
    const float* W    = (const float*)d_in[1];   // [1024,1024]   f32
    float* out        = (float*)d_out;           // [8,2048,1024] f32

    uint8_t* ws = (uint8_t*)d_ws;
    const size_t SZ_REPR = (size_t)BATCH * SEQ * HIDDEN * 2;   // 32 MB bf16
    // layout: reprB | P | reprT | Wt   (~98 MB total, all 16B aligned)
    uint16_t* reprB = (uint16_t*)ws;
    uint16_t* P     = (uint16_t*)(ws + SZ_REPR);
    uint16_t* reprT = (uint16_t*)(ws + 2 * SZ_REPR);
    uint16_t* Wt    = (uint16_t*)(ws + 3 * SZ_REPR);

    cvt_bf16_kernel<<<4096, 256, 0, stream>>>(repr, reprB, BATCH * SEQ * HIDDEN);

    transpose_w_kernel<<<dim3(HIDDEN / 32, HIDDEN / 32), dim3(32, 8), 0, stream>>>(W, Wt);

    transpose_repr_kernel<<<dim3(HIDDEN / 32, SEQ / 32, BATCH), dim3(32, 8), 0, stream>>>(
        repr, reprT);

    dim3 g1(BATCH * SEQ / 64, HIDDEN / 64);      // 256 x 16
    gemm_rw_kernel<<<g1, 128, 0, stream>>>(reprB, Wt, P);

    attn_kernel<<<BATCH * (SEQ / 16), 128, 0, stream>>>(repr, reprB, reprT, P, out);
}
